// sLSTM_72988674228790
// MI455X (gfx1250) — matrix operand order repaired and checked by reading, not compiled, not run on hardware
//
#include <hip/hip_runtime.h>

#define DD   2048
#define BSZ  8192            // B*S rows

typedef __attribute__((ext_vector_type(16))) __bf16 v16bf;
typedef __attribute__((ext_vector_type(8)))  __bf16 v8bf;
typedef __attribute__((ext_vector_type(8)))  float  v8f;

// ---------------- f32 -> bf16 (round-to-nearest-even) conversion ----------------
__device__ __forceinline__ unsigned short f2bf(float f) {
    unsigned int u = __float_as_uint(f);
    u += 0x7FFFu + ((u >> 16) & 1u);
    return (unsigned short)(u >> 16);
}

__global__ __launch_bounds__(256) void cvt_bf16_kernel(const float* __restrict__ in,
                                                       unsigned short* __restrict__ out,
                                                       int n4) {
    int i = blockIdx.x * blockDim.x + threadIdx.x;
    if (i >= n4) return;
    float4 v = ((const float4*)in)[i];
    ushort4 o;
    o.x = f2bf(v.x); o.y = f2bf(v.y); o.z = f2bf(v.z); o.w = f2bf(v.w);
    ((ushort4*)out)[i] = o;
}

// ---------------- WMMA fragment loads (bf16, 16x16x32) ----------------
// A (16x32, MxK), row-major source (row m0+lane%16, ld=DD):
//   lanes 0-15  : VGPR0-3 K=0..7,  VGPR4-7 K=16..23
//   lanes 16-31 : VGPR0-3 K=8..15, VGPR4-7 K=24..31
__device__ __forceinline__ v16bf load_a_frag(const unsigned short* __restrict__ base,
                                             int m0, int k0, int lane) {
    const unsigned short* p = base + (m0 + (lane & 15)) * DD + k0 + ((lane >> 4) << 3);
    v8bf lo = *(const v8bf*)(p);        // K = kh .. kh+7
    v8bf hi = *(const v8bf*)(p + 16);   // K = kh+16 .. kh+23
    return __builtin_shufflevector(lo, hi, 0,1,2,3,4,5,6,7,8,9,10,11,12,13,14,15);
}

// B (32x16, KxN) with B[k][n] = W[n][k] (row-major W, ld=DD):
//   lanes 0-15 hold K=0..7,16..23 for column n=lane; lanes 16-31 hold K=8..15,24..31.
//   Per lane: two 8-bf16 runs from row n of W -> two 16B loads.
__device__ __forceinline__ v16bf load_b_frag(const unsigned short* __restrict__ base,
                                             int n0, int k0, int lane) {
    const unsigned short* p = base + (n0 + (lane & 15)) * DD + k0 + ((lane >> 4) << 3);
    v8bf lo = *(const v8bf*)(p);
    v8bf hi = *(const v8bf*)(p + 16);
    return __builtin_shufflevector(lo, hi, 0,1,2,3,4,5,6,7,8,9,10,11,12,13,14,15);
}

// ---------------- fused 3-gate GEMM + pointwise epilogue ----------------
// Grid: (DD/64, BSZ/128). Block: 256 threads = 8 waves in 4(M) x 2(N).
// Each wave: 32x32 tile per gate, gates {i, c, o} (gate f is dead: f * c_prev == 0).
__global__ __launch_bounds__(256) void slstm_wmma_kernel(
    const unsigned short* __restrict__ xb, const unsigned short* __restrict__ hb,
    const unsigned short* __restrict__ wbi, const unsigned short* __restrict__ wbc,
    const unsigned short* __restrict__ wbo,
    const unsigned short* __restrict__ ubi, const unsigned short* __restrict__ ubc,
    const unsigned short* __restrict__ ubo,
    const float* __restrict__ bWi, const float* __restrict__ bUi,
    const float* __restrict__ bWc, const float* __restrict__ bUc,
    const float* __restrict__ bWo, const float* __restrict__ bUo,
    float* __restrict__ c_out, float* __restrict__ hlast)
{
    const int lane = threadIdx.x & 31;
    const int wave = threadIdx.x >> 5;
    const int m0 = blockIdx.y * 128 + (wave & 3) * 32;
    const int n0 = blockIdx.x * 64  + (wave >> 2) * 32;

    v8f acc[3][4];
    const v8f vzero = {};
    #pragma unroll
    for (int g = 0; g < 3; ++g)
        #pragma unroll
        for (int t = 0; t < 4; ++t) acc[g][t] = vzero;

    #pragma unroll
    for (int ph = 0; ph < 2; ++ph) {
        const unsigned short* A  = ph ? hb  : xb;
        const unsigned short* G0 = ph ? ubi : wbi;
        const unsigned short* G1 = ph ? ubc : wbc;
        const unsigned short* G2 = ph ? ubo : wbo;
        for (int k0 = 0; k0 < DD; k0 += 32) {
            v16bf a0 = load_a_frag(A, m0,      k0, lane);
            v16bf a1 = load_a_frag(A, m0 + 16, k0, lane);
            const unsigned short* Ws[3] = {G0, G1, G2};
            #pragma unroll
            for (int g = 0; g < 3; ++g) {
                v16bf b0 = load_b_frag(Ws[g], n0,      k0, lane);
                v16bf b1 = load_b_frag(Ws[g], n0 + 16, k0, lane);
                acc[g][0] = __builtin_amdgcn_wmma_f32_16x16x32_bf16(false, a0, false, b0, (short)0, acc[g][0], false, false);
                acc[g][1] = __builtin_amdgcn_wmma_f32_16x16x32_bf16(false, a0, false, b1, (short)0, acc[g][1], false, false);
                acc[g][2] = __builtin_amdgcn_wmma_f32_16x16x32_bf16(false, a1, false, b0, (short)0, acc[g][2], false, false);
                acc[g][3] = __builtin_amdgcn_wmma_f32_16x16x32_bf16(false, a1, false, b1, (short)0, acc[g][3], false, false);
            }
        }
    }

    // C/D layout: lane l -> n = l&15; VGPR r -> m = r + (l>>4)*8
    const int cn_lo = lane & 15;
    const int rbase = (lane >> 4) << 3;
    #pragma unroll
    for (int mi = 0; mi < 2; ++mi) {
        #pragma unroll
        for (int ni = 0; ni < 2; ++ni) {
            const int t  = mi * 2 + ni;
            const int cn = n0 + ni * 16 + cn_lo;
            const float bi = bWi[cn] + bUi[cn];
            const float bc = bWc[cn] + bUc[cn];
            const float bo = bWo[cn] + bUo[cn];
            const int mt = m0 + mi * 16 + rbase;
            #pragma unroll
            for (int r = 0; r < 8; ++r) {
                const int cm = mt + r;
                const float zi = acc[0][t][r] + bi;
                const float zc = acc[1][t][r] + bc;
                const float zo = acc[2][t][r] + bo;
                const float iv = __expf(zi);
                const float cv = iv * tanhf(zc);
                c_out[cm * DD + cn] = cv;
                if ((cm & (2048 - 1)) == 2047) {          // s == S-1
                    const float sg = 1.0f / (1.0f + __expf(-zo));
                    hlast[(cm >> 11) * DD + cn] = sg * tanhf(cv);
                }
            }
        }
    }
}

// ---------------- last-timestep projection: hidden[b][e] = hlast[b]·Wp[e] + bp[e] ----------------
__global__ __launch_bounds__(256) void proj_kernel(const float* __restrict__ hlast,
                                                   const float* __restrict__ Wp,
                                                   const float* __restrict__ bp,
                                                   float* __restrict__ out) {
    const int idx = blockIdx.x * blockDim.x + threadIdx.x;   // 8192 threads
    const int b = idx >> 11, e = idx & 2047;
    const float* hr = hlast + b * DD;
    const float* wr = Wp    + e * DD;
    float s = 0.0f;
    for (int d = 0; d < DD; d += 4) {
        float4 hv = *(const float4*)(hr + d);
        float4 wv = *(const float4*)(wr + d);
        s = fmaf(hv.x, wv.x, s); s = fmaf(hv.y, wv.y, s);
        s = fmaf(hv.z, wv.z, s); s = fmaf(hv.w, wv.w, s);
    }
    out[idx] = s + bp[e];
}

// ---------------- workspace layout (bytes) ----------------
#define XB_OFF     0u
#define HB_OFF     33554432u
#define WI_OFF     67108864u
#define WC_OFF     75497472u
#define WO_OFF     83886080u
#define UI_OFF     92274688u
#define UC_OFF     100663296u
#define UO_OFF     109051904u
#define HLAST_OFF  117440512u

extern "C" void kernel_launch(void* const* d_in, const int* in_sizes, int n_in,
                              void* d_out, int out_size, void* d_ws, size_t ws_size,
                              hipStream_t stream) {
    const float* x   = (const float*)d_in[0];
    const float* h   = (const float*)d_in[1];
    const float* Wi  = (const float*)d_in[2];
    const float* bWi = (const float*)d_in[3];
    const float* Ui  = (const float*)d_in[4];
    const float* bUi = (const float*)d_in[5];
    // d_in[6..9] = Wf,bWf,Uf,bUf : mathematically dead (f * c_prev == 0)
    const float* Wc  = (const float*)d_in[10];
    const float* bWc = (const float*)d_in[11];
    const float* Uc  = (const float*)d_in[12];
    const float* bUc = (const float*)d_in[13];
    const float* Wo  = (const float*)d_in[14];
    const float* bWo = (const float*)d_in[15];
    const float* Uo  = (const float*)d_in[16];
    const float* bUo = (const float*)d_in[17];
    const float* Wp  = (const float*)d_in[18];
    const float* bp  = (const float*)d_in[19];

    char* ws = (char*)d_ws;
    unsigned short* xb  = (unsigned short*)(ws + XB_OFF);
    unsigned short* hb  = (unsigned short*)(ws + HB_OFF);
    unsigned short* wbi = (unsigned short*)(ws + WI_OFF);
    unsigned short* wbc = (unsigned short*)(ws + WC_OFF);
    unsigned short* wbo = (unsigned short*)(ws + WO_OFF);
    unsigned short* ubi = (unsigned short*)(ws + UI_OFF);
    unsigned short* ubc = (unsigned short*)(ws + UC_OFF);
    unsigned short* ubo = (unsigned short*)(ws + UO_OFF);
    float*          hl  = (float*)(ws + HLAST_OFF);

    float* out   = (float*)d_out;
    float* c_out = out + 8192;           // tuple order: hidden_state (4x2048) then c

    // f32 -> bf16 conversions (all bf16 data ~117 MB -> resident in 192 MB L2)
    const int big4 = (BSZ * DD) / 4;     // x / h : 16.7M elems
    const int wt4  = (DD * DD) / 4;      // weights: 4.2M elems
    cvt_bf16_kernel<<<big4 / 256, 256, 0, stream>>>(x,  xb,  big4);
    cvt_bf16_kernel<<<big4 / 256, 256, 0, stream>>>(h,  hb,  big4);
    cvt_bf16_kernel<<<wt4 / 256, 256, 0, stream>>>(Wi, wbi, wt4);
    cvt_bf16_kernel<<<wt4 / 256, 256, 0, stream>>>(Wc, wbc, wt4);
    cvt_bf16_kernel<<<wt4 / 256, 256, 0, stream>>>(Wo, wbo, wt4);
    cvt_bf16_kernel<<<wt4 / 256, 256, 0, stream>>>(Ui, ubi, wt4);
    cvt_bf16_kernel<<<wt4 / 256, 256, 0, stream>>>(Uc, ubc, wt4);
    cvt_bf16_kernel<<<wt4 / 256, 256, 0, stream>>>(Uo, ubo, wt4);

    // fused 3-gate WMMA GEMM + activations
    dim3 grid(DD / 64, BSZ / 128);
    slstm_wmma_kernel<<<grid, 256, 0, stream>>>(xb, hb, wbi, wbc, wbo, ubi, ubc, ubo,
                                                bWi, bUi, bWc, bUc, bWo, bUo,
                                                c_out, hl);

    // tiny projection of last timestep
    proj_kernel<<<8192 / 256, 256, 0, stream>>>(hl, Wp, bp, out);
}
